// VimBlock_86217173500355
// MI455X (gfx1250) — matrix operand order repaired and checked by reading, not compiled, not run on hardware
//
#include <hip/hip_runtime.h>
#include <hip/hip_bf16.h>
#include <stdint.h>

// ---------------- problem constants ----------------
#define D_MODEL 256
#define E_DIM   512
#define N_SSM   16
#define BATCH   4
#define SEQ     1024
#define BM_TOK  (BATCH*SEQ)          // 4096 tokens
#define CHUNK   64
#define NCHUNK  (SEQ/CHUNK)          // 16
#define ETILE   16
#define NET     (E_DIM/ETILE)        // 32 e-tiles per direction
#define KT_SCAN (E_DIM/32)           // 16 k-tiles of 32
#define NT_SCAN 33                   // 16 Bm-tiles + 16 Cm-tiles + 1 delta-tile (packed 528 cols)
#define PCOLS   (NT_SCAN*16)         // 528

// ---------------- WMMA types ----------------
typedef __attribute__((ext_vector_type(16))) __bf16 v16bf;
typedef __attribute__((ext_vector_type(8)))  float  v8f;
union Frag { uint32_t u[8]; v16bf v; };

// ---------------- helpers ----------------
__device__ __forceinline__ uint16_t f2bf(float f) {
    uint32_t u = __float_as_uint(f);
    uint32_t r = u + 0x7FFFu + ((u >> 16) & 1u);   // RTNE
    return (uint16_t)(r >> 16);
}
__device__ __forceinline__ float bf2f(uint16_t h) {
    return __uint_as_float(((uint32_t)h) << 16);
}
__device__ __forceinline__ float siluf(float x) {
    return x * (1.f / (1.f + __expf(-x)));
}
__device__ __forceinline__ float softplusf(float x) {
    return (x > 20.f) ? x : log1pf(__expf(x));
}

// ---------------- weight packing: fragment-ready bf16 ----------------
// B-operand fragment layout (16x16x32 bf16): lane l holds column l%16;
// lanes 0-15 carry K 0..15 of the k-tile, lanes 16-31 carry K 16..31;
// u32 v holds {K=2v, K=2v+1}. Stored as dst[frag][lane*8+v] -> each lane
// loads two consecutive uint4 (perfectly coalesced across the wave).
__global__ __launch_bounds__(256) void k_pack_frag(const float* __restrict__ W,
        uint32_t* __restrict__ dst, int K, int N)
{
    int id   = blockIdx.x * 256 + threadIdx.x;
    int idx  = id & 255;
    int frag = id >> 8;
    int lane = idx >> 3, v = idx & 7;
    int Ntiles = N >> 4;
    int kt = frag / Ntiles, nt = frag % Ntiles;
    int c  = nt * 16 + (lane & 15);
    int kb = kt * 32 + ((lane & 16) ? 16 : 0) + 2 * v;
    uint32_t lo = f2bf(W[(size_t)kb * N + c]);
    uint32_t hi = f2bf(W[(size_t)(kb + 1) * N + c]);
    dst[id] = lo | (hi << 16);
}

// Packed per-(e-tile) weight: cols 0..255 = BW(e,n), 256..511 = CW(e,n), 512..527 = DW(e)
__global__ __launch_bounds__(256) void k_pack_scan(const float* __restrict__ BW,
        const float* __restrict__ CW, const float* __restrict__ DW,
        uint32_t* __restrict__ dst)
{
    size_t id = (size_t)blockIdx.x * 256 + threadIdx.x;
    int idx = (int)(id & 255);
    size_t frag = id >> 8;                       // et*KT*NT + kt*NT + nt
    int nt = (int)(frag % NT_SCAN); frag /= NT_SCAN;
    int kt = (int)(frag % KT_SCAN);
    int et = (int)(frag / KT_SCAN);
    int lane = idx >> 3, v = idx & 7;
    int jp = nt * 16 + (lane & 15);
    int kb = kt * 32 + ((lane & 16) ? 16 : 0) + 2 * v;
    float s0, s1;
    if (jp < 256) {
        int c = et * 256 + jp;
        s0 = BW[(size_t)kb * (E_DIM * N_SSM) + c];
        s1 = BW[(size_t)(kb + 1) * (E_DIM * N_SSM) + c];
    } else if (jp < 512) {
        int c = et * 256 + (jp - 256);
        s0 = CW[(size_t)kb * (E_DIM * N_SSM) + c];
        s1 = CW[(size_t)(kb + 1) * (E_DIM * N_SSM) + c];
    } else {
        int c = et * ETILE + (jp - 512);
        s0 = DW[(size_t)kb * E_DIM + c];
        s1 = DW[(size_t)(kb + 1) * E_DIM + c];
    }
    dst[id] = (uint32_t)f2bf(s0) | ((uint32_t)f2bf(s1) << 16);
}

__global__ __launch_bounds__(256) void k_pack_bias(const float* __restrict__ Bb,
        const float* __restrict__ Cb, const float* __restrict__ Db,
        const float* __restrict__ dbias, float* __restrict__ dst)
{
    int id = blockIdx.x * 256 + threadIdx.x;
    if (id >= NET * PCOLS) return;
    int et = id / PCOLS, j = id % PCOLS;
    float v;
    if (j < 256)      v = Bb[et * 256 + j];
    else if (j < 512) v = Cb[et * 256 + (j - 256)];
    else { int e = et * ETILE + (j - 512); v = Db[e] + dbias[e]; }
    dst[id] = v;
}

// ---------------- layernorm -> bf16 ----------------
__global__ __launch_bounds__(256) void k_layernorm(const float* __restrict__ x,
        const float* __restrict__ w, const float* __restrict__ bvec,
        uint16_t* __restrict__ xn)
{
    int lane = threadIdx.x & 31;
    int tok  = blockIdx.x * 8 + (threadIdx.x >> 5);
    const float* xr = x + (size_t)tok * D_MODEL;
    float v[8], s = 0.f, ss = 0.f;
    #pragma unroll
    for (int i = 0; i < 8; ++i) { v[i] = xr[lane + 32 * i]; s += v[i]; ss += v[i] * v[i]; }
    #pragma unroll
    for (int m = 16; m >= 1; m >>= 1) { s += __shfl_xor(s, m, 32); ss += __shfl_xor(ss, m, 32); }
    float mu  = s * (1.f / D_MODEL);
    float var = ss * (1.f / D_MODEL) - mu * mu;
    float rs  = rsqrtf(var + 1e-5f);
    #pragma unroll
    for (int i = 0; i < 8; ++i) {
        int c = lane + 32 * i;
        xn[(size_t)tok * D_MODEL + c] = f2bf((v[i] - mu) * rs * w[c] + bvec[c]);
    }
}

// ---------------- generic WMMA GEMM: out = act(A@W + bias) (+resid) ----------------
// A: (Mrows x K) bf16 row-major; Wf: fragment-packed; 64x64 tile per block, 8 waves.
__global__ __launch_bounds__(256) void k_gemm(const uint16_t* __restrict__ A,
        const uint32_t* __restrict__ Wf, const float* __restrict__ bias,
        const float* __restrict__ resid, float* __restrict__ out,
        int K, int N, int act)
{
    __shared__ uint16_t As[64 * 40];     // 32-wide k panel, +8 pad (bank-conflict free)
    int tid = threadIdx.x, lane = tid & 31, wave = tid >> 5;
    int mt = wave & 3, ng = wave >> 2;
    int mbase = blockIdx.x * 64, nbase = blockIdx.y * 64;
    int Ntiles = N >> 4;
    v8f acc0 = {}, acc1 = {};
    int nkt = K >> 5;
    for (int ks = 0; ks < nkt; ++ks) {
        #pragma unroll
        for (int p = 0; p < 2; ++p) {
            int li = p * 256 + tid;
            int row = li >> 3, seg = li & 7;
            *(uint2*)(As + row * 40 + seg * 4) =
                *(const uint2*)(A + (size_t)(mbase + row) * K + ks * 32 + seg * 4);
        }
        __syncthreads();
        Frag fa;
        #pragma unroll
        for (int v2 = 0; v2 < 8; ++v2) {
            int kl = ((lane & 16) ? 16 : 0) + 2 * v2;
            fa.u[v2] = *(const uint32_t*)(As + (mt * 16 + (lane & 15)) * 40 + kl);
        }
        #pragma unroll
        for (int i = 0; i < 2; ++i) {
            int nt_g = (nbase >> 4) + ng * 2 + i;
            const uint4* wp = (const uint4*)(Wf + ((size_t)ks * Ntiles + nt_g) * 256 + lane * 8);
            Frag fw;
            *(uint4*)&fw.u[0] = wp[0];
            *(uint4*)&fw.u[4] = wp[1];
            if (i == 0)
                acc0 = __builtin_amdgcn_wmma_f32_16x16x32_bf16(false, fa.v, false, fw.v,
                                                               (short)0, acc0, false, false);
            else
                acc1 = __builtin_amdgcn_wmma_f32_16x16x32_bf16(false, fa.v, false, fw.v,
                                                               (short)0, acc1, false, false);
        }
        __syncthreads();
    }
    #pragma unroll
    for (int i = 0; i < 2; ++i) {
        v8f acc = i ? acc1 : acc0;
        int col = nbase + (ng * 2 + i) * 16 + (lane & 15);
        #pragma unroll
        for (int r = 0; r < 8; ++r) {
            int m = mbase + mt * 16 + r + ((lane & 16) ? 8 : 0);
            float val = acc[r] + bias[col];
            if (act == 1) val = siluf(val);
            size_t o = (size_t)m * N + col;
            if (resid) val += resid[o];
            out[o] = val;
        }
    }
}

// ---------------- depthwise conv (K=3, pad=1) + silu -> bf16, both directions ----------------
__global__ __launch_bounds__(256) void k_dwconv(const float* __restrict__ xp,
        const float* __restrict__ wf, const float* __restrict__ bf,
        const float* __restrict__ wb, const float* __restrict__ bb,
        uint16_t* __restrict__ xf, uint16_t* __restrict__ xb)
{
    size_t i = (size_t)blockIdx.x * 256 + threadIdx.x;       // over BM_TOK*E_DIM
    int e = (int)(i % E_DIM);
    size_t tok = i / E_DIM;
    int t = (int)(tok % SEQ);
    float xm = xp[i];
    float xl = (t > 0)       ? xp[i - E_DIM] : 0.f;
    float xr = (t < SEQ - 1) ? xp[i + E_DIM] : 0.f;
    float vf = xl * wf[e * 3 + 0] + xm * wf[e * 3 + 1] + xr * wf[e * 3 + 2] + bf[e];
    float vb = xl * wb[e * 3 + 0] + xm * wb[e * 3 + 1] + xr * wb[e * 3 + 2] + bb[e];
    xf[i] = f2bf(siluf(vf));
    xb[i] = f2bf(siluf(vb));
}

// ---------------- fused chunked projection + selective scan ----------------
// grid: (e-tile, batch, direction). Per chunk: async-copy the 64x512 bf16
// activation chunk global->LDS (ASYNCcnt-tracked, no VGPR round trip), then
// WMMA (64x512)@(512x528) -> LDS, then 64-step recurrence
// (thread = (e_local, n); 16-lane shfl reduction). h persists in registers.
__global__ __launch_bounds__(256) void k_scan(
        const uint16_t* __restrict__ xfp, const uint16_t* __restrict__ xbp,
        const uint32_t* __restrict__ WscanF, const float* __restrict__ biasP,
        const float* __restrict__ A_f, const float* __restrict__ A_b,
        float* __restrict__ yf, float* __restrict__ yb)
{
    extern __shared__ char smem[];
    uint16_t* Alds  = (uint16_t*)smem;                         // 64 x 520 bf16 (pad 8) @ LDS offset 0
    float*    Bmlds = (float*)(smem + 64 * 520 * 2);           // 64 x 256
    float*    Cmlds = Bmlds + 64 * 256;                        // 64 x 256
    float*    dlds  = Cmlds + 64 * 256;                        // 64 x 16 (delta, softplus'ed)
    float*    ylds  = dlds + 64 * 16;                          // 64 x 16

    int tid = threadIdx.x, lane = tid & 31, wave = tid >> 5;
    int et = blockIdx.x, b = blockIdx.y, dir = blockIdx.z;
    const uint16_t* xd   = dir ? xbp : xfp;
    const float*    Amat = dir ? A_b : A_f;
    float*          yout = dir ? yb  : yf;
    const uint32_t* Wb = WscanF + (size_t)(dir * NET + et) * KT_SCAN * NT_SCAN * 256;
    const float*    bp = biasP  + (size_t)(dir * NET + et) * PCOLS;

    int e_loc = tid >> 4, n = tid & 15;
    float A_en = Amat[(size_t)(et * ETILE + e_loc) * N_SSM + n];
    float h = 0.f;

    for (int cc = 0; cc < NCHUNK; ++cc) {
        int chunk = dir ? (NCHUNK - 1 - cc) : cc;
        int t0 = chunk * CHUNK;

        // async-stage 64x512 bf16 activation chunk directly into LDS.
        // Dynamic LDS starts at offset 0 (no static __shared__ here), so the
        // LDS operand is the plain byte offset; 16B aligned (row stride 1040).
        #pragma unroll 4
        for (int p = 0; p < 16; ++p) {
            int li = p * 256 + tid;
            int row = li >> 6, seg = li & 63;
            uint32_t ldsoff = (uint32_t)(row * 1040 + seg * 16);
            uint64_t gaddr  = (uint64_t)(uintptr_t)
                (xd + ((size_t)(b * SEQ + t0 + row)) * E_DIM + seg * 8);
            asm volatile("global_load_async_to_lds_b128 %0, %1, off"
                         :: "v"(ldsoff), "v"(gaddr)
                         : "memory");
        }
        asm volatile("s_wait_asynccnt 0" ::: "memory");
        __syncthreads();

        // WMMA GEMM phase: 4 m-tiles x 33 n-tiles = 132 output tiles over 8 waves
        for (int it = 0; it < 17; ++it) {
            int tile = wave + it * 8;
            if (tile >= 4 * NT_SCAN) break;
            int mt = tile & 3, nt = tile >> 2;
            v8f acc = {};
            for (int ks = 0; ks < KT_SCAN; ++ks) {
                Frag fa;
                #pragma unroll
                for (int v2 = 0; v2 < 8; ++v2) {
                    int kl = ((lane & 16) ? 16 : 0) + 2 * v2;
                    fa.u[v2] = *(const uint32_t*)(Alds + (mt * 16 + (lane & 15)) * 520
                                                  + ks * 32 + kl);
                }
                const uint4* wp = (const uint4*)(Wb + ((size_t)ks * NT_SCAN + nt) * 256 + lane * 8);
                __builtin_prefetch((const char*)wp + NT_SCAN * 256 * 4, 0, 0); // next k-tile
                Frag fw;
                *(uint4*)&fw.u[0] = wp[0];
                *(uint4*)&fw.u[4] = wp[1];
                acc = __builtin_amdgcn_wmma_f32_16x16x32_bf16(false, fa.v, false, fw.v,
                                                              (short)0, acc, false, false);
            }
            // epilogue: bias, route to Bm/Cm/delta LDS, softplus for delta
            int j  = nt * 16 + (lane & 15);
            float bj = bp[j];
            #pragma unroll
            for (int r = 0; r < 8; ++r) {
                int t = mt * 16 + r + ((lane & 16) ? 8 : 0);
                float val = acc[r] + bj;
                if (j >= 512)      dlds[t * 16 + (j - 512)] = softplusf(val);
                else if (j >= 256) Cmlds[t * 256 + (j - 256)] = val;
                else               Bmlds[t * 256 + j] = val;
            }
        }
        __syncthreads();

        // sequential scan over the chunk (reversed for backward direction)
        for (int sstep = 0; sstep < CHUNK; ++sstep) {
            int tl = dir ? (CHUNK - 1 - sstep) : sstep;
            float dlt = dlds[tl * 16 + e_loc];
            float xv  = bf2f(Alds[tl * 520 + et * ETILE + e_loc]);
            float bmv = Bmlds[tl * 256 + e_loc * 16 + n];
            float cmv = Cmlds[tl * 256 + e_loc * 16 + n];
            h = dlt * A_en * h + dlt * bmv * xv;
            float p = h * cmv;
            p += __shfl_xor(p, 1, 32);
            p += __shfl_xor(p, 2, 32);
            p += __shfl_xor(p, 4, 32);
            p += __shfl_xor(p, 8, 32);
            if (n == 0) ylds[tl * 16 + e_loc] = p;
        }
        __syncthreads();

        // flush chunk y to global (block owns this slice exclusively)
        for (int li = tid; li < CHUNK * ETILE; li += 256) {
            int t = li >> 4, e = li & 15;
            yout[((size_t)(b * SEQ + t0 + t)) * E_DIM + et * ETILE + e] = ylds[li];
        }
        __syncthreads();
    }
}

// ---------------- combine: u = (y_f + y_b) * silu(z) -> bf16 ----------------
__global__ __launch_bounds__(256) void k_combine(const float* __restrict__ yf,
        const float* __restrict__ yb, const float* __restrict__ sz,
        uint16_t* __restrict__ u)
{
    size_t i = (size_t)blockIdx.x * 256 + threadIdx.x;
    u[i] = f2bf((yf[i] + yb[i]) * sz[i]);
}

// ---------------- host launch ----------------
extern "C" void kernel_launch(void* const* d_in, const int* in_sizes, int n_in,
                              void* d_out, int out_size, void* d_ws, size_t ws_size,
                              hipStream_t stream)
{
    const float* x      = (const float*)d_in[0];
    const float* norm_w = (const float*)d_in[1];
    const float* norm_b = (const float*)d_in[2];
    const float* Wx  = (const float*)d_in[3];
    const float* bx  = (const float*)d_in[4];
    const float* Wz  = (const float*)d_in[5];
    const float* bz  = (const float*)d_in[6];
    const float* cfw = (const float*)d_in[7];
    const float* cfb = (const float*)d_in[8];
    const float* cbw = (const float*)d_in[9];
    const float* cbb = (const float*)d_in[10];
    const float* BfW = (const float*)d_in[11];
    const float* Bfb = (const float*)d_in[12];
    const float* CfW = (const float*)d_in[13];
    const float* Cfb = (const float*)d_in[14];
    const float* DfW = (const float*)d_in[15];
    const float* Dfb = (const float*)d_in[16];
    const float* dbf = (const float*)d_in[17];
    const float* A_f = (const float*)d_in[18];
    const float* BbW = (const float*)d_in[19];
    const float* Bbb = (const float*)d_in[20];
    const float* CbW = (const float*)d_in[21];
    const float* Cbb = (const float*)d_in[22];
    const float* DbW = (const float*)d_in[23];
    const float* Dbb = (const float*)d_in[24];
    const float* dbb = (const float*)d_in[25];
    const float* A_b  = (const float*)d_in[26];
    const float* Wout = (const float*)d_in[27];
    const float* bout = (const float*)d_in[28];

    char* ws = (char*)d_ws;
    size_t off = 0;
    auto carve = [&](size_t bytes) -> void* {
        off = (off + 255) & ~(size_t)255;
        void* p = ws + off;
        off += bytes;
        return p;
    };
    uint16_t* xn_bf = (uint16_t*)carve((size_t)BM_TOK * D_MODEL * 2);
    float*    xproj = (float*)carve((size_t)BM_TOK * E_DIM * 4);
    float*    sz    = (float*)carve((size_t)BM_TOK * E_DIM * 4);
    uint16_t* xf_bf = (uint16_t*)carve((size_t)BM_TOK * E_DIM * 2);
    uint16_t* xb_bf = (uint16_t*)carve((size_t)BM_TOK * E_DIM * 2);
    float*    yfb   = (float*)carve((size_t)BM_TOK * E_DIM * 4);
    float*    ybb   = (float*)carve((size_t)BM_TOK * E_DIM * 4);
    uint16_t* u_bf  = (uint16_t*)carve((size_t)BM_TOK * E_DIM * 2);
    uint32_t* WxF   = (uint32_t*)carve((size_t)8 * 32 * 256 * 4);
    uint32_t* WzF   = (uint32_t*)carve((size_t)8 * 32 * 256 * 4);
    uint32_t* WoF   = (uint32_t*)carve((size_t)16 * 16 * 256 * 4);
    size_t scanTot  = (size_t)NET * KT_SCAN * NT_SCAN * 256;      // per direction, u32s
    uint32_t* WsF   = (uint32_t*)carve(2 * scanTot * 4);
    float*    bP    = (float*)carve((size_t)2 * NET * PCOLS * 4);

    // weight packing (cheap, fully parallel)
    k_pack_frag<<<(8 * 32 * 256) / 256, 256, 0, stream>>>(Wx, WxF, 256, 512);
    k_pack_frag<<<(8 * 32 * 256) / 256, 256, 0, stream>>>(Wz, WzF, 256, 512);
    k_pack_frag<<<(16 * 16 * 256) / 256, 256, 0, stream>>>(Wout, WoF, 512, 256);
    k_pack_scan<<<(unsigned)(scanTot / 256), 256, 0, stream>>>(BfW, CfW, DfW, WsF);
    k_pack_scan<<<(unsigned)(scanTot / 256), 256, 0, stream>>>(BbW, CbW, DbW, WsF + scanTot);
    k_pack_bias<<<(NET * PCOLS + 255) / 256, 256, 0, stream>>>(Bfb, Cfb, Dfb, dbf, bP);
    k_pack_bias<<<(NET * PCOLS + 255) / 256, 256, 0, stream>>>(Bbb, Cbb, Dbb, dbb, bP + NET * PCOLS);

    // pipeline
    k_layernorm<<<BM_TOK / 8, 256, 0, stream>>>(x, norm_w, norm_b, xn_bf);
    k_gemm<<<dim3(BM_TOK / 64, E_DIM / 64), 256, 0, stream>>>(xn_bf, WxF, bx, nullptr, xproj, 256, 512, 0);
    k_gemm<<<dim3(BM_TOK / 64, E_DIM / 64), 256, 0, stream>>>(xn_bf, WzF, bz, nullptr, sz, 256, 512, 1);
    k_dwconv<<<(BM_TOK * E_DIM) / 256, 256, 0, stream>>>(xproj, cfw, cfb, cbw, cbb, xf_bf, xb_bf);

    size_t scanLds = (size_t)64 * 520 * 2 + (size_t)64 * 256 * 4 * 2 + (size_t)64 * 16 * 4 * 2;
    (void)hipFuncSetAttribute((const void*)k_scan,
                              hipFuncAttributeMaxDynamicSharedMemorySize, (int)scanLds);
    k_scan<<<dim3(NET, BATCH, 2), 256, scanLds, stream>>>(xf_bf, xb_bf, WsF, bP, A_f, A_b, yfb, ybb);

    k_combine<<<(BM_TOK * E_DIM) / 256, 256, 0, stream>>>(yfb, ybb, sz, u_bf);
    k_gemm<<<dim3(BM_TOK / 64, D_MODEL / 64), 256, 0, stream>>>(u_bf, WoF, bout, x, (float*)d_out, 512, 256, 0);
}
